// NeuralSDE_84189948936254
// MI455X (gfx1250) — matrix operand order repaired and checked by reading, not compiled, not run on hardware
//
#include <hip/hip_runtime.h>
#include <hip/hip_bf16.h>

typedef __attribute__((ext_vector_type(16))) __bf16 v16bf;
typedef __attribute__((ext_vector_type(8)))  float  v8f;
typedef unsigned int __attribute__((ext_vector_type(4))) u32x4;   // builtin 128-bit vector (POD)

// explicit global-address-space views (keep global_load after pointer laundering)
typedef __attribute__((address_space(1))) const __bf16 gc_bf16;
typedef __attribute__((address_space(1))) const u32x4  gc_u4;

namespace sde {
constexpr int Bsz = 4096, Tn = 128, Hn = 64, Nn = 16, INn = 32, Wn = 128, Dn = 8;
constexpr int WAVES = 8;
constexpr int ROWS_WG = WAVES * 16;   // 128 batch rows per workgroup
constexpr int NBLK = Bsz / ROWS_WG;   // 32 workgroups

// swizzled bf16 weight blob offsets in d_ws (element units); all stored as
// A-operand tiles of W^T (every GEMM runs transposed: D' = W^T x^T).
constexpr int OFF_DIFF1  = 0;                       // 128x1024 -> A tiles [n rows, K=c]
constexpr int OFF_DIFF0  = OFF_DIFF1 + 128 * 1024;  // 64x128   -> A tiles [c rows, K=h]
constexpr int OFF_DRIFT0 = OFF_DIFF0 + 64 * 128;
constexpr int OFF_DRIFT1 = OFF_DRIFT0 + 64 * 128;   // 128x64   -> A tiles [h rows, K=c]
constexpr int OFF_RO     = OFF_DRIFT1 + 128 * 64;   // 64x16    -> A tiles [d rows, K=h]

// A-operand (16-bit 16xK) interleaved mapping: K = v + 8*hi + 8*(v>=8), lane = hi*16 + row
static __device__ __forceinline__ int swoffA(int k, int n, int numJ) {
  int kc = k >> 5, kk = k & 31;
  int v  = (kk & 7) + ((kk & 16) ? 8 : 0);
  int hi = (kk >> 3) & 1;
  int l  = (hi << 4) + (n & 15);
  return (((kc * numJ) + (n >> 4)) << 9) + (l << 4) + v;
}

// activations on the fast path: v_exp_f32 + v_rcp_f32, no IEEE divide expansion
static __device__ __forceinline__ float fast_tanh(float x) {
  float e = __expf(2.0f * x);
  return 1.0f - 2.0f * __builtin_amdgcn_rcpf(e + 1.0f);
}
static __device__ __forceinline__ float lipswish(float x) {
  return 0.909f * x * __builtin_amdgcn_rcpf(1.0f + __expf(-x));
}
} // namespace sde

// ---------------- prep: weights f32 -> bf16 in WMMA A-operand swizzle ----------------
__global__ void sde_prep_kernel(const float* __restrict__ drift_w0, const float* __restrict__ drift_w1,
                                const float* __restrict__ diff_w0,  const float* __restrict__ diff_w1,
                                const float* __restrict__ ro_w, __bf16* __restrict__ ws) {
  using namespace sde;
  int tid = blockIdx.x * blockDim.x + threadIdx.x;
  int stride = gridDim.x * blockDim.x;
  for (int e = tid; e < 128 * 1024; e += stride) {       // diff_w1 [c, n]
    int k = e >> 10, n = e & 1023;
    ws[OFF_DIFF1 + swoffA(k, n, 64)] = (__bf16)diff_w1[e];
  }
  for (int e = tid; e < 64 * 128; e += stride) {         // layer-0 weights rows 1..64 [h, c]
    int k = e >> 7, n = e & 127;
    ws[OFF_DIFF0  + swoffA(k, n, 8)] = (__bf16)diff_w0[(k + 1) * Wn + n];
    ws[OFF_DRIFT0 + swoffA(k, n, 8)] = (__bf16)drift_w0[(k + 1) * Wn + n];
  }
  for (int e = tid; e < 128 * 64; e += stride) {         // drift_w1 [c, h]
    int k = e >> 6, n = e & 63;
    ws[OFF_DRIFT1 + swoffA(k, n, 4)] = (__bf16)drift_w1[e];
  }
  for (int e = tid; e < 64 * 16; e += stride) {          // ro_w [h, d] zero-padded d: 8->16
    int k = e >> 4, n = e & 15;
    ws[OFF_RO + swoffA(k, n, 1)] = (__bf16)((n < 8) ? ro_w[k * 8 + n] : 0.0f);
  }
}

// ---------------- main: per-wave 16-row ReversibleHeun scan, all GEMMs on WMMA ----------------
__global__ __launch_bounds__(256, 1)
void sde_scan_kernel(const float* __restrict__ tsg,
                     const float* __restrict__ init_noise,
                     const float* __restrict__ dW,
                     const float* __restrict__ init_w0, const float* __restrict__ init_b0,
                     const float* __restrict__ init_w1, const float* __restrict__ init_b1,
                     const float* __restrict__ drift_w0, const float* __restrict__ drift_b0,
                     const float* __restrict__ drift_b1,
                     const float* __restrict__ diff_w0, const float* __restrict__ diff_b0,
                     const float* __restrict__ diff_b1,
                     const float* __restrict__ ro_b,
                     const __bf16* __restrict__ ws,
                     float* __restrict__ out) {
  using namespace sde;
  __shared__ __align__(16) __bf16 s_diff1[128 * 1024];      // 256 KB, resident all 127 steps
  __shared__ __align__(16) __bf16 s_drift1[128 * 64];       // 16 KB
  __shared__ __align__(16) __bf16 s_ro[64 * 16];            // 2 KB
  __shared__ __align__(16) __bf16 s_stage[WAVES][16 * Wn];  // 4 KB/wave staging (per-wave private)
  __shared__ __align__(16) float s_b0d[Wn], s_b0g[Wn], s_wtd[Wn], s_wtg[Wn];
  __shared__ __align__(16) float s_biasd[Wn], s_biasg[Wn];
  __shared__ __align__(16) float s_b1d[Hn], s_b1g[Hn * Nn], s_rob[Dn];

  const int tid  = threadIdx.x;
  const int wave = tid >> 5, lane = tid & 31;
  const int ln = lane & 15, hi = lane >> 4;
  const int rowW = blockIdx.x * ROWS_WG + wave * 16;

  // ---- one-time LDS fill (bulk b128 copy of pre-swizzled blob) ----
  {
    for (int i = tid; i < (128 * 1024) / 8; i += 256)
      *(u32x4*)&s_diff1[i * 8] = *(const u32x4*)(ws + OFF_DIFF1 + i * 8);
    for (int i = tid; i < (128 * 64) / 8; i += 256)
      *(u32x4*)&s_drift1[i * 8] = *(const u32x4*)(ws + OFF_DRIFT1 + i * 8);
    for (int i = tid; i < (64 * 16) / 8; i += 256)
      *(u32x4*)&s_ro[i * 8] = *(const u32x4*)(ws + OFF_RO + i * 8);
    for (int i = tid; i < Wn; i += 256) {
      s_b0d[i] = drift_b0[i]; s_b0g[i] = diff_b0[i];
      s_wtd[i] = drift_w0[i]; s_wtg[i] = diff_w0[i];   // row 0 of w0 = time column
    }
    for (int i = tid; i < Hn; i += 256) s_b1d[i] = drift_b1[i];
    for (int i = tid; i < Hn * Nn; i += 256) s_b1g[i] = diff_b1[i];
    if (tid < Dn) s_rob[tid] = ro_b[tid];
  }
  __syncthreads();

  // ---- LDS accessors: direct indexing of __shared__ arrays (keep addrspace(3)) ----
  auto tileD1 = [&](int kc, int j) -> v16bf {              // diffusion W1 A-tile from LDS
    int e = ((kc * 64 + j) << 9) + (lane << 4);
    union { v16bf v; u32x4 u[2]; } r;
    r.u[0] = *(const u32x4*)&s_diff1[e];
    r.u[1] = *(const u32x4*)&s_diff1[e + 8];
    return r.v;
  };
  auto tileDr1 = [&](int kc, int jt) -> v16bf {            // drift W1 A-tile from LDS
    int e = ((kc * 4 + jt) << 9) + (lane << 4);
    union { v16bf v; u32x4 u[2]; } r;
    r.u[0] = *(const u32x4*)&s_drift1[e];
    r.u[1] = *(const u32x4*)&s_drift1[e + 8];
    return r.v;
  };
  auto tileRO = [&](int kc) -> v16bf {                     // readout A-tile from LDS
    int e = (kc << 9) + (lane << 4);
    union { v16bf v; u32x4 u[2]; } r;
    r.u[0] = *(const u32x4*)&s_ro[e];
    r.u[1] = *(const u32x4*)&s_ro[e + 8];
    return r.v;
  };
  auto loadBrm = [&](int ldk, int kc) -> v16bf {           // B-operand from staging [16 b][ldk K]
    int e = ln * ldk + kc * 32 + hi * 8;
    union { v16bf v; u32x4 u[2]; } r;
    r.u[0] = *(const u32x4*)&s_stage[wave][e];
    r.u[1] = *(const u32x4*)&s_stage[wave][e + 16];
    return r.v;
  };
  auto stageStore = [&](const float* v, int ldk, int c0) { // pack 8 f32->bf16, one b128 store
    union { u32x4 u; __bf16 h[8]; } pk;
#pragma unroll
    for (int k = 0; k < 8; ++k) pk.h[k] = (__bf16)v[k];
    *(u32x4*)&s_stage[wave][ln * ldk + c0] = pk.u;
  };
  auto tileG = [&](gc_bf16* blob, int numJ, int kc, int j) -> v16bf {
    union { v16bf v; u32x4 u[2]; } r;                      // global pre-swizzled A-tile (L2-resident)
    gc_u4* q = (gc_u4*)(blob + (((kc * numJ + j) << 9) + (lane << 4)));
    r.u[0] = q[0]; r.u[1] = q[1];
    return r.v;
  };
  auto mma = [](v16bf a, v16bf b, v8f c) -> v8f {
    return __builtin_amdgcn_wmma_f32_16x16x32_bf16(false, a, false, b, (short)0, c, false, false);
  };

  // Evaluate both nets at (t, xs). Tiles transposed [feature, batch]:
  // lane holds (row = k + 8*hi, col b = ln). Overwrites fO/qO in place;
  // accumulates 0.5 * (g . dwi) directly into yAcc when doUpd.
  auto evalStep = [&](float t, int i, bool doUpd,
                      v8f (&xs)[4], v8f (&fO)[4], v8f (&qO)[4], v8f (&yAcc)[4]) {
    // Launder layer-0 weight base pointers (addrspace(1) so loads stay
    // global_load): opaque per-step redefinition stops LICM from hoisting
    // the tile loads out of the scan loop (previously caused scratch spills).
    gc_bf16* wd0 = (gc_bf16*)(ws + OFF_DRIFT0);
    gc_bf16* wg0 = (gc_bf16*)(ws + OFF_DIFF0);
    asm volatile("" : "+s"(wd0), "+s"(wg0));

    __syncthreads();   // previous step's bias reads are done
    // per-step folded biases: b0 + t * w_t (cross-wave, once per step)
    if (tid < Wn) s_biasd[tid] = s_b0d[tid] + t * s_wtd[tid];
    else          s_biasg[tid - Wn] = s_b0g[tid - Wn] + t * s_wtg[tid - Wn];
    // stage x^T as row-major [16 b][64 h] (per-wave slice; same-wave DS order suffices)
#pragma unroll
    for (int nt = 0; nt < 4; ++nt) {
      float v[8];
#pragma unroll
      for (int k = 0; k < 8; ++k) v[k] = xs[nt][k];
      stageStore(v, 64, nt * 16 + 8 * hi);
    }
    __syncthreads();   // bias vectors ready (staging needs no barrier)
    v16bf bx0 = loadBrm(64, 0), bx1 = loadBrm(64, 1);

    // ---- drift layer 0 (transposed): h' = lipswish(W0^T x^T + bias) ----
#pragma unroll
    for (int j = 0; j < 8; ++j) {
      v8f acc = {0.f,0.f,0.f,0.f,0.f,0.f,0.f,0.f};
      acc = mma(tileG(wd0, 8, 0, j), bx0, acc);
      acc = mma(tileG(wd0, 8, 1, j), bx1, acc);
      int c0 = j * 16 + 8 * hi;
      float bv[8], hv[8];
      *(float4*)(bv)     = *(const float4*)&s_biasd[c0];
      *(float4*)(bv + 4) = *(const float4*)&s_biasd[c0 + 4];
#pragma unroll
      for (int k = 0; k < 8; ++k) hv[k] = lipswish(acc[k] + bv[k]);
      stageStore(hv, Wn, c0);
      if (j & 1) __builtin_amdgcn_sched_barrier(0);   // pair tiles: hide WMMA->VALU hazard
    }
    v16bf bh[4];
#pragma unroll
    for (int kc = 0; kc < 4; ++kc) bh[kc] = loadBrm(Wn, kc);

    // ---- drift layer 1 (transposed, LDS weights): f' = tanh(W1^T h^T + b1) ----
#pragma unroll
    for (int jt = 0; jt < 4; ++jt) {
      v8f acc = {0.f,0.f,0.f,0.f,0.f,0.f,0.f,0.f};
#pragma unroll
      for (int kc = 0; kc < 4; ++kc)
        acc = mma(tileDr1(kc, jt), bh[kc], acc);
      int c0 = jt * 16 + 8 * hi;
      float bv[8];
      *(float4*)(bv)     = *(const float4*)&s_b1d[c0];
      *(float4*)(bv + 4) = *(const float4*)&s_b1d[c0 + 4];
#pragma unroll
      for (int k = 0; k < 8; ++k) fO[jt][k] = fast_tanh(acc[k] + bv[k]);
      if (jt & 1) __builtin_amdgcn_sched_barrier(0);
    }

    // Brownian increments: issue loads here so they overlap diffusion layer 0
    float dwiT[8], dwnT[8];
    {
      const size_t rowOff = (size_t)(rowW + ln) * (Tn - 1) * Nn + 8 * hi;
      const int ii = (i < 0) ? 0 : i;
      const float* dwp = dW + rowOff + (size_t)ii * Nn;
      *(float4*)(dwiT)     = *(const float4*)(dwp);
      *(float4*)(dwiT + 4) = *(const float4*)(dwp + 4);
      if (i + 1 < Tn - 1) {
        const float* dnp = dW + rowOff + (size_t)(i + 1) * Nn;
        *(float4*)(dwnT)     = *(const float4*)(dnp);
        *(float4*)(dwnT + 4) = *(const float4*)(dnp + 4);
      } else {
#pragma unroll
        for (int k = 0; k < 8; ++k) dwnT[k] = 0.0f;
      }
    }

    // ---- diffusion layer 0 (transposed, reuse staging) ----
#pragma unroll
    for (int j = 0; j < 8; ++j) {
      v8f acc = {0.f,0.f,0.f,0.f,0.f,0.f,0.f,0.f};
      acc = mma(tileG(wg0, 8, 0, j), bx0, acc);
      acc = mma(tileG(wg0, 8, 1, j), bx1, acc);
      int c0 = j * 16 + 8 * hi;
      float bv[8], hv[8];
      *(float4*)(bv)     = *(const float4*)&s_biasg[c0];
      *(float4*)(bv + 4) = *(const float4*)&s_biasg[c0 + 4];
#pragma unroll
      for (int k = 0; k < 8; ++k) hv[k] = lipswish(acc[k] + bv[k]);
      stageStore(hv, Wn, c0);
      if (j & 1) __builtin_amdgcn_sched_barrier(0);
    }
#pragma unroll
    for (int kc = 0; kc < 4; ++kc) bh[kc] = loadBrm(Wn, kc);

    // ---- diffusion layer 1 (transposed) with fused dW projections ----
    // Tile j = one h index: lane holds g[b=ln, h=j, n=k+8*hi].
    // In-lane dot over 8 n-values, one xor-16 combine; q replaced in place and
    // 0.5*p accumulated straight into y at static slot (j&7) when hi matches.
#pragma unroll
    for (int jt = 0; jt < 4; ++jt) {
#pragma unroll
      for (int jj = 0; jj < 16; ++jj) {
        const int j = jt * 16 + jj;
        v8f acc = {0.f,0.f,0.f,0.f,0.f,0.f,0.f,0.f};
#pragma unroll
        for (int kc = 0; kc < 4; ++kc)
          acc = mma(tileD1(kc, j), bh[kc], acc);
        int c0 = j * 16 + 8 * hi;
        float bv[8];
        *(float4*)(bv)     = *(const float4*)&s_b1g[c0];
        *(float4*)(bv + 4) = *(const float4*)&s_b1g[c0 + 4];
        float pp = 0.f, qq = 0.f;
#pragma unroll
        for (int k = 0; k < 8; ++k) {
          float gv = fast_tanh(acc[k] + bv[k]);
          pp = fmaf(gv, dwiT[k], pp);
          qq = fmaf(gv, dwnT[k], qq);
        }
        pp += __shfl_xor(pp, 16, 32);
        qq += __shfl_xor(qq, 16, 32);
        if (hi == ((jj >> 3) & 1)) {
          qO[jt][jj & 7] = qq;
          if (doUpd) yAcc[jt][jj & 7] += 0.5f * pp;
        }
        if (jj & 1) __builtin_amdgcn_sched_barrier(0);
      }
    }
  };

  // readout (transposed): out[b, t, :] = ro_w^T y^T + ro_b ; lane hi==0 owns row b=ln
  auto readout = [&](v8f (&ys)[4], int t) {
#pragma unroll
    for (int nt = 0; nt < 4; ++nt) {
      float v[8];
#pragma unroll
      for (int k = 0; k < 8; ++k) v[k] = ys[nt][k];
      stageStore(v, 64, nt * 16 + 8 * hi);
    }
    v16bf b0 = loadBrm(64, 0), b1 = loadBrm(64, 1);
    v8f acc = {0.f,0.f,0.f,0.f,0.f,0.f,0.f,0.f};
    acc = mma(tileRO(0), b0, acc);
    acc = mma(tileRO(1), b1, acc);
    if (hi == 0) {
      float rv[8], ov[8];
      *(float4*)(rv)     = *(const float4*)&s_rob[0];
      *(float4*)(rv + 4) = *(const float4*)&s_rob[4];
#pragma unroll
      for (int k = 0; k < 8; ++k) ov[k] = acc[k] + rv[k];
      float* op = out + ((size_t)(rowW + ln) * Tn + t) * Dn;
      *(float4*)(op)     = *(float4*)(ov);
      *(float4*)(op + 4) = *(float4*)(ov + 4);
    }
  };

  // ---- initial MLP (one-time, plain VALU): y0 = relu(x@W0+b0)@W1+b1 ----
  {
    const float* xr = init_noise + (size_t)(rowW + ln) * INn;
    float xv[INn];
#pragma unroll
    for (int k = 0; k < INn; ++k) xv[k] = xr[k];
    for (int cc = 0; cc < 64; ++cc) {
      int c = hi * 64 + cc;
      float acc = init_b0[c];
      for (int k = 0; k < INn; ++k) acc = fmaf(xv[k], init_w0[k * Wn + c], acc);
      s_stage[wave][ln * Wn + c] = (__bf16)fmaxf(acc, 0.0f);
    }
  }
  __syncthreads();
  float yac[32];
#pragma unroll
  for (int ii = 0; ii < 32; ++ii) {
    int idx = ii * 32 + lane;
    int r = idx >> 6, h2 = idx & 63;
    float acc = init_b1[h2];
    for (int c = 0; c < Wn; ++c)
      acc = fmaf((float)s_stage[wave][r * Wn + c], init_w1[c * Hn + h2], acc);
    yac[ii] = acc;
  }
  __syncthreads();
  float* fst = (float*)&s_stage[wave][0];   // alias staging as [16 b][64 h] f32
#pragma unroll
  for (int ii = 0; ii < 32; ++ii) fst[ii * 32 + lane] = yac[ii];
  __syncthreads();

  // load y0 into transposed tiles: lane holds (h = nt*16 + k + 8*hi, b = ln)
  v8f y[4], yh[4], f[4], q[4];
#pragma unroll
  for (int nt = 0; nt < 4; ++nt) {
    float bv[8];
    *(float4*)(bv)     = *(const float4*)(fst + ln * 64 + nt * 16 + 8 * hi);
    *(float4*)(bv + 4) = *(const float4*)(fst + ln * 64 + nt * 16 + 8 * hi + 4);
#pragma unroll
    for (int k = 0; k < 8; ++k) {
      y[nt][k] = bv[k]; yh[nt][k] = bv[k];
      f[nt][k] = 0.0f;  q[nt][k] = 0.0f;   // i = -1 bootstrap
    }
  }

  // ---- unified scan: i = -1 bootstraps f0/q0, i >= 0 are ReversibleHeun steps ----
  for (int i = -1; i < Tn - 1; ++i) {
    float tb = tsg[i + 1];
    float dt = (i >= 0) ? (tb - tsg[i]) : 0.0f;
    const bool doUpd = (i >= 0);
    // u = f*dt + q ; yh <- 2y - yh + u (the new yhat) ; y pre-accumulates 0.5*u
#pragma unroll
    for (int nt = 0; nt < 4; ++nt) {
      v8f u = f[nt] * dt + q[nt];
      yh[nt] = 2.0f * y[nt] - yh[nt] + u;
      y[nt]  = y[nt] + 0.5f * u;        // u == 0 at i == -1
    }
    evalStep(tb, i, doUpd, yh, f, q, y);  // overwrites f,q; adds 0.5*p into y
    if (doUpd) {
#pragma unroll
      for (int nt = 0; nt < 4; ++nt) y[nt] = y[nt] + (0.5f * dt) * f[nt];
    }
    readout(y, i + 1);
  }
}

extern "C" void kernel_launch(void* const* d_in, const int* in_sizes, int n_in,
                              void* d_out, int out_size, void* d_ws, size_t ws_size,
                              hipStream_t stream) {
  (void)in_sizes; (void)n_in; (void)out_size; (void)ws_size;
  const float* tsg        = (const float*)d_in[0];
  const float* init_noise = (const float*)d_in[1];
  const float* dW         = (const float*)d_in[2];
  const float* init_w0    = (const float*)d_in[3];
  const float* init_b0    = (const float*)d_in[4];
  const float* init_w1    = (const float*)d_in[5];
  const float* init_b1    = (const float*)d_in[6];
  const float* drift_w0   = (const float*)d_in[7];
  const float* drift_b0   = (const float*)d_in[8];
  const float* drift_w1   = (const float*)d_in[9];
  const float* drift_b1   = (const float*)d_in[10];
  const float* diff_w0    = (const float*)d_in[11];
  const float* diff_b0    = (const float*)d_in[12];
  const float* diff_w1    = (const float*)d_in[13];
  const float* diff_b1    = (const float*)d_in[14];
  const float* ro_w       = (const float*)d_in[15];
  const float* ro_b       = (const float*)d_in[16];
  __bf16* ws = (__bf16*)d_ws;
  float* out = (float*)d_out;

  sde_prep_kernel<<<256, 256, 0, stream>>>(drift_w0, drift_w1, diff_w0, diff_w1, ro_w, ws);
  sde_scan_kernel<<<sde::NBLK, sde::WAVES * 32, 0, stream>>>(
      tsg, init_noise, dW, init_w0, init_b0, init_w1, init_b1,
      drift_w0, drift_b0, drift_b1, diff_w0, diff_b0, diff_b1, ro_b, ws, out);
}
